// GNN_25254407700575
// MI455X (gfx1250) — matrix operand order repaired and checked by reading, not compiled, not run on hardware
//
#include <hip/hip_runtime.h>
#include <cstdint>
#include <cstddef>

#define D 128

typedef unsigned short u16;
typedef __bf16 bf16_t;
typedef bf16_t v16bf __attribute__((ext_vector_type(16)));
typedef bf16_t v8bf  __attribute__((ext_vector_type(8)));
typedef float  v8f   __attribute__((ext_vector_type(8)));
typedef u16    v4u16 __attribute__((ext_vector_type(4)));

__device__ __forceinline__ u16 f2bf(float f) {
  unsigned u = __float_as_uint(f);
  unsigned r = u + 0x7FFFu + ((u >> 16) & 1u);   // round-to-nearest-even
  return (u16)(r >> 16);
}
__device__ __forceinline__ float bf2f(u16 h) {
  return __uint_as_float(((unsigned)h) << 16);
}
__device__ __forceinline__ void atomicMaxF(float* a, float v) {
  unsigned old = __float_as_uint(*a);
  while (__uint_as_float(old) < v) {
    unsigned prev = atomicCAS((unsigned*)a, old, __float_as_uint(v));
    if (prev == old) break;
    old = prev;
  }
}

// ---------------------------------------------------------------------------
// Prep: build the 12 GEMM B-matrices (bf16, WMMA-fragment order) + folded q bias
//   kind 0: M = qw^T @ W   (folds qW = q@W into a single GEMM), plus bq' = qb@W
//   kind 1: M = kw^T        kind 2: M = vw^T
// Fragment order: idx = ((t8*4+c)*32 + lane)*16 + i,
//   o = t8*16 + (lane&15),  k = c*32 + (i<8 ? (lane>>4)*8 + i : 16 + (lane>>4)*8 + (i-8))
// so each B fragment in the GEMM is one contiguous 32B load per lane.
// ---------------------------------------------------------------------------
__global__ void prep_weights(const float* __restrict__ qwu, const float* __restrict__ qwi,
                             const float* __restrict__ kwu, const float* __restrict__ kwi,
                             const float* __restrict__ vwu, const float* __restrict__ vwi,
                             const float* __restrict__ Wui, const float* __restrict__ Wiu,
                             const float* __restrict__ qbu, const float* __restrict__ qbi,
                             u16* __restrict__ wfrag, float* __restrict__ bqc) {
  int b = blockIdx.x;                 // 0..11
  int l = b / 6, t = (b / 3) % 2, kind = b % 3;
  const float* Wm = (t == 0 ? Wui : Wiu) + l * D * D;
  u16* out = wfrag + (size_t)b * D * D;
  for (int idx = threadIdx.x; idx < D * D; idx += 256) {
    int i = idx & 15, lane = (idx >> 4) & 31, c = (idx >> 9) & 3, t8 = idx >> 11;
    int o = t8 * 16 + (lane & 15);
    int k = c * 32 + (i < 8 ? (lane >> 4) * 8 + i : 16 + (lane >> 4) * 8 + (i - 8));
    float val;
    if (kind == 0) {
      const float* qw = (t == 0 ? qwu : qwi) + l * D * D;
      float s = 0.f;
      for (int j = 0; j < D; ++j) s += qw[j * D + k] * Wm[j * D + o];
      val = s;
    } else if (kind == 1) {
      const float* kw = (t == 0 ? kwi : kwu) + l * D * D;  // dst-type weights
      val = kw[o * D + k];
    } else {
      const float* vw = (t == 0 ? vwu : vwi) + l * D * D;  // src-type weights
      val = vw[o * D + k];
    }
    out[idx] = f2bf(val);
  }
  if (kind == 0 && threadIdx.x < D) {
    const float* qb = (t == 0 ? qbu : qbi) + l * D;
    int o = threadIdx.x;
    float s = 0.f;
    for (int j = 0; j < D; ++j) s += qb[j] * Wm[j * D + o];
    bqc[(l * 2 + t) * D + o] = s;
  }
}

__global__ void cvt_f32_bf16(const float* __restrict__ in, u16* __restrict__ out, int n) {
  for (int i = blockIdx.x * blockDim.x + threadIdx.x; i < n; i += gridDim.x * blockDim.x)
    out[i] = f2bf(in[i]);
}

// ---------------------------------------------------------------------------
// WMMA GEMM: Y[N,128](bf16) = X[N,128](bf16) @ M[128,128] + bias
// Block: 256 thr = 8 waves, 128 rows staged into LDS; wave = 16 rows,
// 8 column tiles x 4 K-chunks = 32 v_wmma_f32_16x16x32_bf16 per wave.
// ---------------------------------------------------------------------------
#define LP 136   // padded LDS row pitch (bf16 elements)

__global__ void __launch_bounds__(256) gemm_bf16(const u16* __restrict__ X,
                                                 const u16* __restrict__ Bfrag,
                                                 const float* __restrict__ bias,
                                                 u16* __restrict__ Y, int N) {
  __shared__ u16 xs[128 * LP];
  const int tid = threadIdx.x;
  const int rowBlk = blockIdx.x * 128;
  // cooperative coalesced stage: 128 rows x 64 dwords
  for (int idx = tid; idx < 128 * 64; idx += 256) {
    int r = idx >> 6, cu = idx & 63;
    int gr = rowBlk + r;
    unsigned val = 0u;
    if (gr < N) val = ((const unsigned*)X)[(size_t)gr * 64 + cu];
    *(unsigned*)&xs[r * LP + cu * 2] = val;
  }
  __syncthreads();

  const int wave = tid >> 5, lane = tid & 31;
  const int rowBase = rowBlk + wave * 16;
  if (rowBase >= N) return;
  const int m = lane & 15, h = lane >> 4;

  const u16* rowp = xs + (wave * 16 + m) * LP;
  v16bf a[4];
#pragma unroll
  for (int c = 0; c < 4; ++c) {
    // 16-bit A 16x32 layout: lane half h: elems 0..7 -> K = 32c + 8h .. +7,
    //                                    elems 8..15 -> K = 32c + 16 + 8h .. +7
    v8bf lo = *(const v8bf*)(rowp + c * 32 + h * 8);
    v8bf hi = *(const v8bf*)(rowp + c * 32 + 16 + h * 8);
    a[c] = __builtin_shufflevector(lo, hi, 0, 1, 2, 3, 4, 5, 6, 7,
                                   8, 9, 10, 11, 12, 13, 14, 15);
  }

  const v16bf* Bf = (const v16bf*)Bfrag;
#pragma unroll
  for (int t8 = 0; t8 < 8; ++t8) {
    v8f cacc = {};
#pragma unroll
    for (int c = 0; c < 4; ++c) {
      v16bf bfr = Bf[(t8 * 4 + c) * 32 + lane];
      cacc = __builtin_amdgcn_wmma_f32_16x16x32_bf16(
          false, a[c], false, bfr, (short)0, cacc, false, false);
    }
    const int n = t8 * 16 + m;
    const float bv = bias[n];
#pragma unroll
    for (int r = 0; r < 8; ++r) {
      int row = rowBase + h * 8 + r;   // C/D layout: VGPR r -> M = 8h + r
      Y[(size_t)row * D + n] = f2bf(cacc[r] + bv);
    }
  }
}

__global__ void init_scal(float* scal) { scal[0] = -3.402823466e38f; scal[1] = 0.f; }

// wave-per-edge bilinear score + leaky_relu, global max via block reduce
__global__ void edge_score(const u16* __restrict__ q, const u16* __restrict__ k,
                           const int* __restrict__ src, const int* __restrict__ dst,
                           float* __restrict__ s, float* __restrict__ scal, int E) {
  const int lane = threadIdx.x & 31;
  const int gw = (blockIdx.x * blockDim.x + threadIdx.x) >> 5;
  const int nw = (gridDim.x * blockDim.x) >> 5;
  float lmax = -3.402823466e38f;
  for (int e = gw; e < E; e += nw) {
    int si = src[e], di = dst[e];
    v4u16 qa = *(const v4u16*)(q + (size_t)si * D + lane * 4);
    v4u16 ka = *(const v4u16*)(k + (size_t)di * D + lane * 4);
    float acc = 0.f;
#pragma unroll
    for (int j = 0; j < 4; ++j) acc += bf2f(qa[j]) * bf2f(ka[j]);
#pragma unroll
    for (int off = 16; off; off >>= 1) acc += __shfl_xor(acc, off, 32);
    acc *= 0.08838834764831845f;                 // 1/sqrt(128)
    acc = (acc > 0.f) ? acc : acc * 0.01f;       // leaky_relu
    if (lane == 0) s[e] = acc;
    lmax = fmaxf(lmax, acc);
  }
  __shared__ float red[8];
  int wv = threadIdx.x >> 5;
  if (lane == 0) red[wv] = lmax;
  __syncthreads();
  if (threadIdx.x == 0) {
    float mm = red[0];
    for (int i = 1; i < (int)(blockDim.x >> 5); ++i) mm = fmaxf(mm, red[i]);
    atomicMaxF(scal, mm);
  }
}

__global__ void edge_expsum(float* __restrict__ s, float* __restrict__ scal, int E) {
  const float m = scal[0];
  float lsum = 0.f;
  for (int i = blockIdx.x * blockDim.x + threadIdx.x; i < E; i += gridDim.x * blockDim.x) {
    float t = __expf(s[i] - m);
    s[i] = t;
    lsum += t;
  }
#pragma unroll
  for (int off = 16; off; off >>= 1) lsum += __shfl_xor(lsum, off, 32);
  __shared__ float red[8];
  int lane = threadIdx.x & 31, wv = threadIdx.x >> 5;
  if (lane == 0) red[wv] = lsum;
  __syncthreads();
  if (threadIdx.x == 0) {
    float t = 0.f;
    for (int i = 0; i < (int)(blockDim.x >> 5); ++i) t += red[i];
    atomicAdd(scal + 1, t);
  }
}

// wave-per-edge: acc[dst] += (exp/sum) * v[src]   (f32 atomics into L2)
__global__ void edge_scatter(const float* __restrict__ s, const u16* __restrict__ v,
                             const int* __restrict__ src, const int* __restrict__ dst,
                             const float* __restrict__ scal, float* __restrict__ acc, int E) {
  const float inv = 1.0f / scal[1];
  const int lane = threadIdx.x & 31;
  const int gw = (blockIdx.x * blockDim.x + threadIdx.x) >> 5;
  const int nw = (gridDim.x * blockDim.x) >> 5;
  for (int e = gw; e < E; e += nw) {
    float alpha = s[e] * inv;
    int si = src[e], di = dst[e];
    v4u16 va = *(const v4u16*)(v + (size_t)si * D + lane * 4);
    float* ap = acc + (size_t)di * D + lane * 4;
#pragma unroll
    for (int j = 0; j < 4; ++j) atomicAdd(ap + j, alpha * bf2f(va[j]));
  }
}

// ---------------------------------------------------------------------------
extern "C" void kernel_launch(void* const* d_in, const int* in_sizes, int n_in,
                              void* d_out, int out_size, void* d_ws, size_t ws_size,
                              hipStream_t stream) {
  (void)n_in; (void)out_size; (void)ws_size;
  const float* x_user = (const float*)d_in[0];
  const float* x_item = (const float*)d_in[1];
  const int*   ei_ui  = (const int*)d_in[2];
  const int*   ei_iu  = (const int*)d_in[3];
  const float* qwu = (const float*)d_in[4];
  const float* qbu = (const float*)d_in[5];
  const float* kwu = (const float*)d_in[6];
  const float* kbu = (const float*)d_in[7];
  const float* vwu = (const float*)d_in[8];
  const float* vbu = (const float*)d_in[9];
  const float* qwi = (const float*)d_in[10];
  const float* qbi = (const float*)d_in[11];
  const float* kwi = (const float*)d_in[12];
  const float* kbi = (const float*)d_in[13];
  const float* vwi = (const float*)d_in[14];
  const float* vbi = (const float*)d_in[15];
  const float* Wui = (const float*)d_in[16];
  const float* Wiu = (const float*)d_in[17];

  const int NU = in_sizes[0] / D;
  const int NI = in_sizes[1] / D;
  const int E  = in_sizes[2] / 2;
  const int NDu = NU * D, NDi = NI * D;
  const int Nmax = NU > NI ? NU : NI;

  // ---- workspace layout (256B aligned chunks) ----
  char* w = (char*)d_ws;
  size_t off = 0;
  auto alloc = [&](size_t bytes) -> char* {
    char* p = w + off;
    off += (bytes + 255) & ~(size_t)255;
    return p;
  };
  u16* xu16  = (u16*)alloc((size_t)NDu * 2);          // bf16 layer input (user)
  u16* xi16  = (u16*)alloc((size_t)NDi * 2);          // bf16 layer input (item)
  u16* q16   = (u16*)alloc((size_t)Nmax * D * 2);     // qW projection (bf16)
  u16* k16   = (u16*)alloc((size_t)Nmax * D * 2);     // k projection   (bf16)
  u16* v16   = (u16*)alloc((size_t)Nmax * D * 2);     // v projection   (bf16)
  u16* wfrag = (u16*)alloc((size_t)12 * D * D * 2);   // 12 fragment-ordered matrices
  float* bqc = (float*)alloc((size_t)4 * D * 4);      // folded q biases
  float* scal = (float*)alloc(16);                    // [max, sum]
  float* sbuf = (float*)alloc((size_t)E * 4);         // edge scores / exp
  float* x1   = (float*)alloc((size_t)(NDu + NDi) * 4); // layer-1 f32 accumulators

  prep_weights<<<12, 256, 0, stream>>>(qwu, qwi, kwu, kwi, vwu, vwi,
                                       Wui, Wiu, qbu, qbi, wfrag, bqc);

  const int cgrid = 2048;
  hipMemcpyAsync(x1, x_user, (size_t)NDu * 4, hipMemcpyDeviceToDevice, stream);
  hipMemcpyAsync(x1 + NDu, x_item, (size_t)NDi * 4, hipMemcpyDeviceToDevice, stream);
  cvt_f32_bf16<<<cgrid, 256, 0, stream>>>(x_user, xu16, NDu);
  cvt_f32_bf16<<<cgrid, 256, 0, stream>>>(x_item, xi16, NDi);

  auto run_agg = [&](int l, int t, const u16* s16, const u16* d16, int Ns, int Nd,
                     const int* ei, const float* kb, const float* vb, float* acc) {
    int base = (l * 2 + t) * 3;
    int gb_s = (Ns + 127) / 128, gb_d = (Nd + 127) / 128;
    gemm_bf16<<<gb_s, 256, 0, stream>>>(s16, wfrag + (size_t)(base + 0) * D * D,
                                        bqc + (l * 2 + t) * D, q16, Ns);
    gemm_bf16<<<gb_d, 256, 0, stream>>>(d16, wfrag + (size_t)(base + 1) * D * D,
                                        kb, k16, Nd);
    gemm_bf16<<<gb_s, 256, 0, stream>>>(s16, wfrag + (size_t)(base + 2) * D * D,
                                        vb, v16, Ns);
    init_scal<<<1, 1, 0, stream>>>(scal);
    edge_score<<<1024, 256, 0, stream>>>(q16, k16, ei, ei + E, sbuf, scal, E);
    edge_expsum<<<1024, 256, 0, stream>>>(sbuf, scal, E);
    edge_scatter<<<2048, 256, 0, stream>>>(sbuf, v16, ei, ei + E, scal, acc, E);
  };

  // ---- layer 0 ----
  run_agg(0, 0, xu16, xi16, NU, NI, ei_ui, kbi + 0, vbu + 0, x1 + NDu); // user->item
  run_agg(0, 1, xi16, xu16, NI, NU, ei_iu, kbu + 0, vbi + 0, x1);       // item->user

  // ---- layer 1 ----
  cvt_f32_bf16<<<cgrid, 256, 0, stream>>>(x1, xu16, NDu);
  cvt_f32_bf16<<<cgrid, 256, 0, stream>>>(x1 + NDu, xi16, NDi);
  float* outp = (float*)d_out;
  hipMemcpyAsync(outp, x1, (size_t)(NDu + NDi) * 4, hipMemcpyDeviceToDevice, stream);
  run_agg(1, 0, xu16, xi16, NU, NI, ei_ui, kbi + D, vbu + D, outp + NDu);
  run_agg(1, 1, xi16, xu16, NI, NU, ei_iu, kbu + D, vbi + D, outp);
}
